// DenseGCN_35734127903505
// MI455X (gfx1250) — matrix operand order repaired and checked by reading, compiled, not run on hardware
//
#include <hip/hip_runtime.h>

#define B_   64
#define N_   512
#define F_   4
#define IN_  64
#define H_   128
#define OUT_ 16

typedef __attribute__((ext_vector_type(16))) _Float16 v16h;
typedef __attribute__((ext_vector_type(8)))  _Float16 v8h;
typedef __attribute__((ext_vector_type(8)))  float    v8f;

__device__ __forceinline__ float silu_f(float x) {
  return x / (1.f + __expf(-x));
}

// ---------------------------------------------------------------------------
// K1: degree -> dn[b,i,f] = rsqrt(max(1, sum_j adj'[b,i,j,f])), self-loop=1
// one block per (b,i), 256 threads, float4 over the F=4 edge types
// ---------------------------------------------------------------------------
__global__ __launch_bounds__(256) void k_deg(const float* __restrict__ adj,
                                             float* __restrict__ dn) {
  int bi = blockIdx.x;            // b*512 + i
  int i  = bi & (N_ - 1);
  const float4* arow = (const float4*)adj + (size_t)bi * N_;
  float4 s = make_float4(0.f, 0.f, 0.f, 0.f);
  for (int j = threadIdx.x; j < N_; j += 256) {
    float4 a = arow[j];
    if (j == i) { a.x = 1.f; a.y = 1.f; a.z = 1.f; a.w = 1.f; }
    s.x += a.x; s.y += a.y; s.z += a.z; s.w += a.w;
  }
  __shared__ float4 red[256];
  red[threadIdx.x] = s;
  __syncthreads();
  for (int off = 128; off > 0; off >>= 1) {
    if ((int)threadIdx.x < off) {
      float4 o = red[threadIdx.x + off];
      red[threadIdx.x].x += o.x; red[threadIdx.x].y += o.y;
      red[threadIdx.x].z += o.z; red[threadIdx.x].w += o.w;
    }
    __syncthreads();
  }
  if (threadIdx.x == 0) {
    float4 t = red[0];
    float4 d;
    d.x = rsqrtf(fmaxf(t.x, 1.f));
    d.y = rsqrtf(fmaxf(t.y, 1.f));
    d.z = rsqrtf(fmaxf(t.z, 1.f));
    d.w = rsqrtf(fmaxf(t.w, 1.f));
    ((float4*)dn)[bi] = d;
  }
}

// ---------------------------------------------------------------------------
// K2: A2[b,i,j] = sum_f dn[b,i,f]*adj'[b,i,j,f]*dn[b,j,f]  (f16 out)
// ---------------------------------------------------------------------------
__global__ __launch_bounds__(256) void k_a2(const float* __restrict__ adj,
                                            const float* __restrict__ dn,
                                            _Float16* __restrict__ A2) {
  size_t t = (size_t)blockIdx.x * 256 + threadIdx.x;   // b*N*N + i*N + j
  if (t >= (size_t)B_ * N_ * N_) return;
  int    j  = (int)(t & (N_ - 1));
  size_t bi = t >> 9;                                  // b*N + i
  int    i  = (int)(bi & (N_ - 1));
  size_t b  = bi >> 9;
  float4 a  = ((const float4*)adj)[t];
  if (j == i) { a.x = 1.f; a.y = 1.f; a.z = 1.f; a.w = 1.f; }
  float4 di = ((const float4*)dn)[bi];
  float4 dj = ((const float4*)dn)[(b << 9) + j];
  float v = di.x * a.x * dj.x + di.y * a.y * dj.y +
            di.z * a.z * dj.z + di.w * a.w * dj.w;
  A2[t] = (_Float16)v;
}

// ---------------------------------------------------------------------------
// f32 -> f16 copy (x), and small weight transpose W(KxH) -> Wt(HxK) f16
// ---------------------------------------------------------------------------
__global__ __launch_bounds__(256) void k_f16copy(const float* __restrict__ in,
                                                 _Float16* __restrict__ out,
                                                 int n) {
  int t = blockIdx.x * 256 + threadIdx.x;
  if (t < n) out[t] = (_Float16)in[t];
}

__global__ __launch_bounds__(256) void k_wt(const float* __restrict__ W,
                                            _Float16* __restrict__ Wt,
                                            int K, int H) {
  int t = blockIdx.x * 256 + threadIdx.x;
  if (t >= K * H) return;
  int c = t / K, k = t - c * K;
  Wt[t] = (_Float16)W[k * H + c];    // Wt[c][k] = W[k][c]
}

// ---------------------------------------------------------------------------
// Unified NT WMMA GEMM:  D(MxN) = A(MxK) * S^T,  A,S row-major f16 (K contig)
// per wave: 16(M) x 64(N) tile, 4 x v8f accumulators, K-loop step 32.
// mode 0: store f16 D                     (dense layers; D layout = next S)
// mode 1: silu((D+bias[n])*mask[m]) f16   (GCN layer 0)
// mode 2: silu((D+bias[n])*mask[m]) f32   (GCN layer 1)
// ---------------------------------------------------------------------------
__global__ __launch_bounds__(128) void k_gemm_nt(
    const _Float16* __restrict__ A, const _Float16* __restrict__ S,
    void* __restrict__ D, const float* __restrict__ bias,
    const int* __restrict__ mask,
    int M, int N, int K,
    long long aStride, long long sStride, long long dStride, int mode) {
  int lane = threadIdx.x & 31;
  int wid  = blockIdx.x * 4 + (threadIdx.x >> 5);
  int mTiles = M >> 4, nStrips = N >> 6;
  int perB = mTiles * nStrips;
  int b  = wid / perB;
  int r  = wid - b * perB;
  int mt = r / nStrips, ns = r - mt * nStrips;
  int m0 = mt << 4, n0 = ns << 6;

  const _Float16* Ab = A + (size_t)b * aStride;
  const _Float16* Sb = S + (size_t)b * sStride;

  int hi = lane >> 4;          // half-wave id
  int ml = lane & 15;
  // A fragment: row m0+ml, 16 contiguous halves at k0+16*hi
  const _Float16* aRow = Ab + (size_t)(m0 + ml) * K + (hi << 4);
  // S (=B^T) fragment: row n0+nt*16+ml, 16 contiguous halves at k0+16*hi
  const _Float16* sRow = Sb + (size_t)(n0 + ml) * K + (hi << 4);

  v8f acc[4] = {};
  for (int k0 = 0; k0 < K; k0 += 32) {
    v8h alo = *(const v8h*)(aRow + k0);
    v8h ahi = *(const v8h*)(aRow + k0 + 8);
    v16h af = __builtin_shufflevector(alo, ahi, 0, 1, 2, 3, 4, 5, 6, 7,
                                      8, 9, 10, 11, 12, 13, 14, 15);
#pragma unroll
    for (int nt = 0; nt < 4; ++nt) {
      v16h bf = *(const v16h*)(sRow + (size_t)(nt * 16) * K + k0);
      acc[nt] = __builtin_amdgcn_wmma_f32_16x16x32_f16(
          false, af, false, bf, (short)0, acc[nt], false, false);
    }
  }

  size_t dOff  = (size_t)b * dStride;
  int    mBase = m0 + (hi << 3);
#pragma unroll
  for (int nt = 0; nt < 4; ++nt) {
    int   n  = n0 + nt * 16 + ml;
    float bv = (mode != 0) ? bias[n] : 0.f;
#pragma unroll
    for (int v = 0; v < 8; ++v) {
      int   m   = mBase + v;
      float val = acc[nt][v];
      if (mode != 0) {
        float mf = (float)mask[(size_t)b * M + m];
        val = (val + bv) * mf;
        val = silu_f(val);
      }
      if (mode == 2)
        ((float*)D)[dOff + (size_t)m * N + n] = val;
      else
        ((_Float16*)D)[dOff + (size_t)m * N + n] = (_Float16)val;
    }
  }
}

// ---------------------------------------------------------------------------
// K6: mean-pool over nodes + 2-layer MLP head. one block per batch.
// ---------------------------------------------------------------------------
__global__ __launch_bounds__(128) void k_pool(const float* __restrict__ h1,
                                              const float* __restrict__ Wl1,
                                              const float* __restrict__ bl1,
                                              const float* __restrict__ Wl2,
                                              const float* __restrict__ bl2,
                                              float* __restrict__ out) {
  int b = blockIdx.x;
  int c = threadIdx.x;              // 0..127
  const float* hb = h1 + (size_t)b * N_ * H_;
  float s = 0.f;
  for (int i = 0; i < N_; ++i) s += hb[(size_t)i * H_ + c];
  __shared__ float g[H_];
  __shared__ float gl[H_];
  g[c] = s * (1.f / (float)N_);
  __syncthreads();
  float a = bl1[c];
  for (int k = 0; k < H_; ++k) a += g[k] * Wl1[k * H_ + c];
  gl[c] = silu_f(a);
  __syncthreads();
  if (c < OUT_) {
    float o = bl2[c];
    for (int k = 0; k < H_; ++k) o += gl[k] * Wl2[k * OUT_ + c];
    out[b * OUT_ + c] = o;
  }
}

// ---------------------------------------------------------------------------
extern "C" void kernel_launch(void* const* d_in, const int* in_sizes, int n_in,
                              void* d_out, int out_size, void* d_ws, size_t ws_size,
                              hipStream_t stream) {
  const float* x    = (const float*)d_in[0];
  const float* adj  = (const float*)d_in[1];
  const int*   mask = (const int*)d_in[2];
  const float* W0   = (const float*)d_in[3];
  const float* b0   = (const float*)d_in[4];
  const float* W1   = (const float*)d_in[5];
  const float* b1   = (const float*)d_in[6];
  const float* Wl1  = (const float*)d_in[7];
  const float* bl1  = (const float*)d_in[8];
  const float* Wl2  = (const float*)d_in[9];
  const float* bl2  = (const float*)d_in[10];

  char* ws = (char*)d_ws;
  float*    dn   = (float*)(ws);                 //   524,288 B  (B*N*F f32)
  _Float16* A2   = (_Float16*)(ws + 524288);     // 33,554,432 B (B*N*N f16)
  _Float16* xf16 = (_Float16*)(ws + 34078720);   //  4,194,304 B
  _Float16* w0t  = (_Float16*)(ws + 38273024);   //     16,384 B
  _Float16* w1t  = (_Float16*)(ws + 38289408);   //     32,768 B
  _Float16* bt   = (_Float16*)(ws + 38322176);   //  8,388,608 B (xw_t / h1w_t)
  _Float16* h0   = (_Float16*)(ws + 46710784);   //  8,388,608 B
  float*    h1   = (float*)(ws + 55099392);      // 16,777,216 B  -> total ~71.9 MB

  // 1) degree norm
  k_deg<<<B_ * N_, 256, 0, stream>>>(adj, dn);
  // 2) fused normalized adjacency (f16)
  k_a2<<<(B_ * N_ * N_) / 256, 256, 0, stream>>>(adj, dn, A2);
  // 3) f16 prep
  k_f16copy<<<(B_ * N_ * IN_) / 256, 256, 0, stream>>>(x, xf16, B_ * N_ * IN_);
  k_wt<<<(IN_ * H_ + 255) / 256, 256, 0, stream>>>(W0, w0t, IN_, H_);
  k_wt<<<(H_ * H_ + 255) / 256, 256, 0, stream>>>(W1, w1t, H_, H_);
  // 4) xw_t[b][c][j] = (x @ W0)^T : M=128(c), N=512(j), K=64
  k_gemm_nt<<<1024, 128, 0, stream>>>(w0t, xf16, bt, nullptr, nullptr,
                                      H_, N_, IN_, 0, (long long)N_ * IN_,
                                      (long long)H_ * N_, 0);
  // 5) h0[b][i][c] = silu((A2 @ xw + b0) * mask) : M=512, N=128, K=512
  k_gemm_nt<<<1024, 128, 0, stream>>>(A2, bt, h0, b0, mask,
                                      N_, H_, N_, (long long)N_ * N_,
                                      (long long)H_ * N_, (long long)N_ * H_, 1);
  // 6) h1w_t[b][c][j] = (h0 @ W1)^T : M=128, N=512, K=128
  k_gemm_nt<<<1024, 128, 0, stream>>>(w1t, h0, bt, nullptr, nullptr,
                                      H_, N_, H_, 0, (long long)N_ * H_,
                                      (long long)H_ * N_, 0);
  // 7) h1[b][i][c] = silu((A2 @ h1w + b1) * mask), fp32 out
  k_gemm_nt<<<1024, 128, 0, stream>>>(A2, bt, h1, b1, mask,
                                      N_, H_, N_, (long long)N_ * N_,
                                      (long long)H_ * N_, (long long)N_ * H_, 2);
  // 8) mean pool + MLP head
  k_pool<<<B_, 128, 0, stream>>>(h1, Wl1, bl1, Wl2, bl2, (float*)d_out);
}
